// PointNetFeaturePropagation_77068893160406
// MI455X (gfx1250) — compile-verified, matrix operand order checked by reading, not run on hardware
//
#include <hip/hip_runtime.h>
#include <stdint.h>

// ---------------------------------------------------------------------------
// PointNet Feature Propagation for MI455X (gfx1250, wave32, WMMA)
//   B=16, N=4096, S=1024, D1=D2=512, C_in=1024, H1=512, H2=256
// fp32 3-NN interpolation -> bf16 staging -> two WMMA bf16 GEMMs (f32 accum,
// fused bias+ReLU). Weight tiles stream to LDS via gfx1250 async-to-LDS
// (ASYNCcnt), double-buffered; A fragments software-pipelined with a 2-step
// unroll (ping-pong register sets -> no copy movs, no WMMA hazard NOPs).
// ---------------------------------------------------------------------------

typedef __attribute__((ext_vector_type(16))) __bf16 v16bf;
typedef __attribute__((ext_vector_type(8)))  __bf16 v8bf;
typedef __attribute__((ext_vector_type(8)))  float  v8f;

#define B_SZ   16
#define N_PTS  4096
#define S_PTS  1024
#define D1_C   512
#define D2_C   512
#define CIN_C  1024
#define H1_C   512
#define H2_C   256
#define M_ROWS (B_SZ * N_PTS)          // 65536
#define EPSV   1e-8f

// ---------------------------------------------------------------------------
// Async 16B copy global -> LDS (gfx1250 GLOBAL_LOAD_ASYNC_TO_LDS_B128).
// For generic pointers in the LDS aperture, addr[31:0] is the LDS byte
// address, which is exactly what the VDST operand wants.
// ---------------------------------------------------------------------------
__device__ __forceinline__ void async_copy_b128(__bf16* lds_dst,
                                                const __bf16* gsrc) {
  uint32_t lds_addr = (uint32_t)(uintptr_t)lds_dst;
  asm volatile("global_load_async_to_lds_b128 %0, %1, off"
               :: "v"(lds_addr), "v"(gsrc)
               : "memory");
}
__device__ __forceinline__ void wait_async0() {
  asm volatile("s_wait_asynccnt 0x0" ::: "memory");
}

// ---------------------------------------------------------------------------
// Kernel 1: convert + transpose weights to bf16 [Nout, K] so that WMMA
// B-fragments become contiguous 16B loads per lane.
// ---------------------------------------------------------------------------
__global__ __launch_bounds__(256) void transpose_cvt_kernel(
    const float* __restrict__ W, __bf16* __restrict__ Wt, int K, int Nc) {
  int idx = blockIdx.x * 256 + threadIdx.x;
  if (idx < K * Nc) {
    int k = idx / Nc;
    int n = idx - k * Nc;
    Wt[(size_t)n * K + k] = (__bf16)W[idx];
  }
}

// ---------------------------------------------------------------------------
// Kernel 2: 3-NN inverse-distance interpolation + concat, bf16 output.
// One block = 256 query points of one batch; xyz2[b] staged in LDS (12 KB).
// ---------------------------------------------------------------------------
__global__ __launch_bounds__(256) void interp_concat_kernel(
    const float* __restrict__ xyz1, const float* __restrict__ xyz2,
    const float* __restrict__ points1, const float* __restrict__ points2,
    __bf16* __restrict__ X) {
  __shared__ float sx[S_PTS];
  __shared__ float sy[S_PTS];
  __shared__ float sz[S_PTS];

  const int b     = blockIdx.x >> 4;
  const int chunk = blockIdx.x & 15;
  const int t     = threadIdx.x;

  const float* x2 = xyz2 + (size_t)b * S_PTS * 3;
  for (int i = t; i < S_PTS; i += 256) {
    sx[i] = x2[i * 3 + 0];
    sy[i] = x2[i * 3 + 1];
    sz[i] = x2[i * 3 + 2];
  }
  __syncthreads();

  const int    n   = chunk * 256 + t;
  const size_t row = (size_t)b * N_PTS + n;
  const float  px  = xyz1[row * 3 + 0];
  const float  py  = xyz1[row * 3 + 1];
  const float  pz  = xyz1[row * 3 + 2];

  float d0 = 3.4e38f, d1 = 3.4e38f, d2 = 3.4e38f;
  int   i0 = 0, i1 = 0, i2 = 0;
  for (int s = 0; s < S_PTS; ++s) {
    float dx = px - sx[s], dy = py - sy[s], dz = pz - sz[s];
    float d  = dx * dx + dy * dy + dz * dz;
    if (d < d0)      { d2 = d1; i2 = i1; d1 = d0; i1 = i0; d0 = d; i0 = s; }
    else if (d < d1) { d2 = d1; i2 = i1; d1 = d;  i1 = s; }
    else if (d < d2) { d2 = d;  i2 = s; }
  }
  const float r0 = 1.f / (d0 + EPSV);
  const float r1 = 1.f / (d1 + EPSV);
  const float r2 = 1.f / (d2 + EPSV);
  const float rs = 1.f / (r0 + r1 + r2);
  const float w0 = r0 * rs, w1 = r1 * rs, w2 = r2 * rs;

  __bf16*      Xr  = X + row * CIN_C;
  const float* p1r = points1 + row * D1_C;
  const float* q0  = points2 + ((size_t)b * S_PTS + i0) * D2_C;
  const float* q1  = points2 + ((size_t)b * S_PTS + i1) * D2_C;
  const float* q2  = points2 + ((size_t)b * S_PTS + i2) * D2_C;

  for (int c = 0; c < D1_C; c += 4) {
    float4 a = *(const float4*)(p1r + c);
    Xr[c + 0] = (__bf16)a.x;  Xr[c + 1] = (__bf16)a.y;
    Xr[c + 2] = (__bf16)a.z;  Xr[c + 3] = (__bf16)a.w;
  }
  for (int c = 0; c < D2_C; c += 4) {
    float4 f0 = *(const float4*)(q0 + c);
    float4 f1 = *(const float4*)(q1 + c);
    float4 f2 = *(const float4*)(q2 + c);
    Xr[D1_C + c + 0] = (__bf16)(w0 * f0.x + w1 * f1.x + w2 * f2.x);
    Xr[D1_C + c + 1] = (__bf16)(w0 * f0.y + w1 * f1.y + w2 * f2.y);
    Xr[D1_C + c + 2] = (__bf16)(w0 * f0.z + w1 * f1.z + w2 * f2.z);
    Xr[D1_C + c + 3] = (__bf16)(w0 * f0.w + w1 * f1.w + w2 * f2.w);
  }
}

// ---------------------------------------------------------------------------
// Kernel 3/4: C = relu(A[M,KD](bf16) * Bt[NC,KD](bf16)^T + bias).
// Block tile 128x128; 8 waves as 4(M) x 2(N); wave tile 32x64 = 2x4 WMMAs.
// ---------------------------------------------------------------------------
struct GemmCtx {
  const __bf16* arow[2];   // per-lane A row base (k index 0 at fragment kb0)
  const __bf16* btcol;     // Bt + blockN-relative per-thread staging source
  __bf16*       sB0;       // LDS buffer 0 (this thread's staging dst slot)
  __bf16*       sB1;       // LDS buffer 1
  const __bf16* sBr0;      // LDS buffer 0 fragment read base (wave slot)
  const __bf16* sBr1;      // LDS buffer 1 fragment read base
  int kb0;
};

// One pipelined k-step: waits tile kt, prefetches tile kt+1 into the other
// LDS buffer, loads A fragments for kt+1 into `aout`, does 8 WMMAs with `ain`.
template <int KD>
__device__ __forceinline__ void gemm_step(const GemmCtx& c, int kkn,
                                          __bf16* stage_dst,
                                          const __bf16* frag_src,
                                          const v16bf (&ain)[2],
                                          v16bf (&aout)[2],
                                          v8f (&acc)[2][4]) {
  wait_async0();          // our async writes for tile kt have landed
  __syncthreads();        // everyone's have; prior readers of dst are done

  // async-prefetch next B tile (32 k) into the other buffer
#pragma unroll
  for (int p = 0; p < 2; ++p)
    async_copy_b128(stage_dst + p * (64 * 40),
                    c.btcol + (size_t)p * 64 * KD + kkn);

  // pipeline next A fragments into the *other* register set (no copies later)
#pragma unroll
  for (int i = 0; i < 2; ++i) {
    v8bf lo = *(const v8bf*)(c.arow[i] + kkn);
    v8bf hi = *(const v8bf*)(c.arow[i] + kkn + 16);
#pragma unroll
    for (int e = 0; e < 8; ++e) { aout[i][e] = lo[e]; aout[i][e + 8] = hi[e]; }
  }

  // B fragments for this k-step from LDS
  v16bf bfrag[4];
#pragma unroll
  for (int j = 0; j < 4; ++j) {
    const __bf16* br = frag_src + j * (16 * 40);
    v8bf lo = *(const v8bf*)(br);
    v8bf hi = *(const v8bf*)(br + 16);
#pragma unroll
    for (int e = 0; e < 8; ++e) { bfrag[j][e] = lo[e]; bfrag[j][e + 8] = hi[e]; }
  }

#pragma unroll
  for (int i = 0; i < 2; ++i)
#pragma unroll
    for (int j = 0; j < 4; ++j)
      acc[i][j] = __builtin_amdgcn_wmma_f32_16x16x32_bf16(
          false, ain[i], false, bfrag[j], (short)0, acc[i][j], false, false);
}

template <bool OUT_BF16, int NC, int KD>
__global__ __launch_bounds__(256) void gemm_bias_relu_kernel(
    const __bf16* __restrict__ A, const __bf16* __restrict__ Bt,
    const float* __restrict__ bias, void* __restrict__ Cout) {
  // padded row stride 40 bf16 (80 B): 16B-aligned chunks, spread banks
  __shared__ __attribute__((aligned(16))) __bf16 sB[2][128 * 40];

  const int t    = threadIdx.x;
  const int lane = t & 31;
  const int wid  = t >> 5;
  const int wm   = wid >> 1;           // 0..3  (M strip of 32 rows)
  const int wn   = wid & 1;            // 0..1  (N strip of 64 cols)
  const int blockN = blockIdx.x * 128;
  const int blockM = blockIdx.y * 128;

  // ISA 16-bit fragment layout: lanes 0-15 -> K 0..7 (lo) / 16..23 (hi),
  // lanes 16-31 -> K 8..15 (lo) / 24..31 (hi); row/col = lane & 15.
  const int kb0  = (lane < 16) ? 0 : 8;
  const int mrow = lane & 15;

  const int chunk = t & 3;   // 16B k-chunk within a 32-k row
  const int colh  = t >> 2;  // 0..63

  GemmCtx c;
  c.kb0 = kb0;
#pragma unroll
  for (int i = 0; i < 2; ++i)
    c.arow[i] = A + (size_t)(blockM + wm * 32 + i * 16 + mrow) * KD + kb0;
  c.btcol = Bt + (size_t)(blockN + colh) * KD + chunk * 8;
  c.sB0   = &sB[0][colh * 40 + chunk * 8];
  c.sB1   = &sB[1][colh * 40 + chunk * 8];
  c.sBr0  = &sB[0][(wn * 64 + mrow) * 40 + kb0];
  c.sBr1  = &sB[1][(wn * 64 + mrow) * 40 + kb0];

  v8f acc[2][4];
#pragma unroll
  for (int i = 0; i < 2; ++i)
#pragma unroll
    for (int j = 0; j < 4; ++j)
      acc[i][j] = (v8f){0.f, 0.f, 0.f, 0.f, 0.f, 0.f, 0.f, 0.f};

  // ---- prologue: async-prefetch B tile 0, preload A fragments for k=0 ----
#pragma unroll
  for (int p = 0; p < 2; ++p)
    async_copy_b128(c.sB0 + p * (64 * 40), c.btcol + (size_t)p * 64 * KD);

  v16bf aA[2], aB[2];
#pragma unroll
  for (int i = 0; i < 2; ++i) {
    v8bf lo = *(const v8bf*)(c.arow[i]);
    v8bf hi = *(const v8bf*)(c.arow[i] + 16);
#pragma unroll
    for (int e = 0; e < 8; ++e) { aA[i][e] = lo[e]; aA[i][e + 8] = hi[e]; }
  }

  // ---- main loop, unrolled by 2: ping-pong LDS buffers AND A registers ----
  constexpr int NT = KD / 32;          // 32 or 16 (even)
  for (int kt = 0; kt < NT; kt += 2) {
    const int kkn0 = (kt + 1 < NT) ? (kt + 1) * 32 : kt * 32;
    gemm_step<KD>(c, kkn0, c.sB1, c.sBr0, aA, aB, acc);
    const int kkn1 = (kt + 2 < NT) ? (kt + 2) * 32 : kkn0;
    gemm_step<KD>(c, kkn1, c.sB0, c.sBr1, aB, aA, acc);
  }

  // Epilogue: bias + ReLU. C layout: col = lane&15, row = vgpr + 8*(lane>=16).
  const int mhalf = (lane >> 4) * 8;
#pragma unroll
  for (int i = 0; i < 2; ++i) {
#pragma unroll
    for (int j = 0; j < 4; ++j) {
      const int   col = blockN + wn * 64 + j * 16 + (lane & 15);
      const float bv  = bias[col];
      const size_t base =
          (size_t)(blockM + wm * 32 + i * 16 + mhalf) * NC + col;
#pragma unroll
      for (int r = 0; r < 8; ++r) {
        float v = acc[i][j][r] + bv;
        v = v > 0.f ? v : 0.f;
        if (OUT_BF16)
          ((__bf16*)Cout)[base + (size_t)r * NC] = (__bf16)v;
        else
          ((float*)Cout)[base + (size_t)r * NC] = v;
      }
    }
  }
}

// ---------------------------------------------------------------------------
extern "C" void kernel_launch(void* const* d_in, const int* in_sizes, int n_in,
                              void* d_out, int out_size, void* d_ws,
                              size_t ws_size, hipStream_t stream) {
  const float* xyz1    = (const float*)d_in[0];
  const float* xyz2    = (const float*)d_in[1];
  const float* points1 = (const float*)d_in[2];
  const float* points2 = (const float*)d_in[3];
  const float* W1      = (const float*)d_in[4];
  const float* b1      = (const float*)d_in[5];
  const float* W2      = (const float*)d_in[6];
  const float* b2      = (const float*)d_in[7];
  float* out = (float*)d_out;

  // Workspace layout:
  //   Xbf : 65536 x 1024 bf16 = 128 MiB @ 0
  //   Hbf : 65536 x  512 bf16 =  64 MiB @ 128 MiB
  //   W1t :   512 x 1024 bf16 =   1 MiB @ 192 MiB
  //   W2t :   256 x  512 bf16 = 256 KiB @ 193 MiB
  char*   ws  = (char*)d_ws;
  __bf16* Xbf = (__bf16*)(ws);
  __bf16* Hbf = (__bf16*)(ws + ((size_t)128 << 20));
  __bf16* W1t = (__bf16*)(ws + ((size_t)192 << 20));
  __bf16* W2t = (__bf16*)(ws + ((size_t)193 << 20));

  transpose_cvt_kernel<<<(CIN_C * H1_C + 255) / 256, 256, 0, stream>>>(
      W1, W1t, CIN_C, H1_C);
  transpose_cvt_kernel<<<(H1_C * H2_C + 255) / 256, 256, 0, stream>>>(
      W2, W2t, H1_C, H2_C);

  interp_concat_kernel<<<B_SZ * (N_PTS / 256), 256, 0, stream>>>(
      xyz1, xyz2, points1, points2, Xbf);

  {
    dim3 grid(H1_C / 128, M_ROWS / 128);
    gemm_bias_relu_kernel<true, H1_C, CIN_C><<<grid, 256, 0, stream>>>(
        Xbf, W1t, b1, (void*)Hbf);
  }
  {
    dim3 grid(H2_C / 128, M_ROWS / 128);
    gemm_bias_relu_kernel<false, H2_C, H1_C><<<grid, 256, 0, stream>>>(
        Hbf, W2t, b2, (void*)out);
  }
}
